// NearestEmbed_58523224375336
// MI455X (gfx1250) — matrix operand order repaired and checked by reading, not compiled, not run on hardware
//
#include <hip/hip_runtime.h>

typedef __attribute__((ext_vector_type(16))) _Float16 v16h;
typedef __attribute__((ext_vector_type(8)))  float    v8f;

#define Dch   128
#define Kcb   1024
#define HW    4096
#define NB    32
#define NPOS  (NB * HW)        // 131072 positions
#define NTILE (NPOS / 16)      // 8192 16-row tiles

// workspace layout (bytes)
#define WS_PACKED_OFF 0        // 131072 f16  = 262144 B : WMMA-swizzled B tiles
#define WS_NE2_OFF    262144   // 1024  f32   =   4096 B : -0.5*||e||^2
#define WS_WT_OFF     266240   // 131072 f32  = 524288 B : codebook transposed [k][d]

// ---------------------------------------------------------------------------
// Prep: f16 B-tiles in exact per-lane WMMA layout, -||e||^2/2, and W^T (f32).
// B layout (16-bit 32x16): lane holds N = lane&15; halves h=0..15 hold
// K = (h<8 ? 0 : 16) + (lane>=16 ? 8 : 0) + (h&7).  Packed index = ((c*4+t)*32 + lane)*16 + h.
// ---------------------------------------------------------------------------
__global__ void vq_prep_kernel(const float* __restrict__ w,
                               _Float16* __restrict__ packed,
                               float* __restrict__ ne2,
                               float* __restrict__ wT) {
  const int i = blockIdx.x * blockDim.x + threadIdx.x;  // 0..131071
  {
    const int h   = i & 15;
    const int L   = (i >> 4) & 31;
    const int t   = (i >> 9) & 3;
    const int c   = i >> 11;
    const int n   = c * 16 + (L & 15);
    const int sel = L >> 4;
    const int d   = t * 32 + (h >> 3) * 16 + sel * 8 + (h & 7);
    packed[i] = (_Float16)w[d * Kcb + n];
  }
  {
    const int k = i >> 7;
    const int d = i & 127;
    wT[i] = w[d * Kcb + k];
  }
  if (i < Kcb) {
    float s = 0.f;
    for (int d = 0; d < Dch; ++d) { const float v = w[d * Kcb + i]; s += v * v; }
    ne2[i] = -0.5f * s;
  }
}

// ---------------------------------------------------------------------------
// helpers
// ---------------------------------------------------------------------------
__device__ __forceinline__ void load_btile(const v16h* __restrict__ Bp, int c,
                                           int lane, v16h B[4]) {
#pragma unroll
  for (int t = 0; t < 4; ++t) B[t] = Bp[(c * 4 + t) * 32 + lane];
}

__device__ __forceinline__ void mma_step(const v16h A0[4], const v16h A1[4],
                                         const v16h B[4], float c0, int n,
                                         float bV0[8], int bI0[8],
                                         float bV1[8], int bI1[8]) {
  v8f a0 = {};   // zero-init C -> inline 0 SRC2
  v8f a1 = {};
#pragma unroll
  for (int t = 0; t < 4; ++t) {   // two independent 4-deep chains, interleaved
    a0 = __builtin_amdgcn_wmma_f32_16x16x32_f16(false, A0[t], false, B[t],
                                                (short)0, a0, false, false);
    a1 = __builtin_amdgcn_wmma_f32_16x16x32_f16(false, A1[t], false, B[t],
                                                (short)0, a1, false, false);
  }
#pragma unroll
  for (int r = 0; r < 8; ++r) {
    const float v0 = a0[r] + c0;
    if (v0 > bV0[r]) { bV0[r] = v0; bI0[r] = n; }
    const float v1 = a1[r] + c0;
    if (v1 > bV1[r]) { bV1[r] = v1; bI1[r] = n; }
  }
}

// ---------------------------------------------------------------------------
// Main: one wave per PAIR of adjacent 16-position tiles (M=32 per wave).
// 64 code tiles x 8 wmma(16x16x32 f16), B double-buffered across iterations.
// ---------------------------------------------------------------------------
__global__ void __launch_bounds__(256) vq_main_kernel(
    const float* __restrict__ x,
    const _Float16* __restrict__ packed,
    const float* __restrict__ ne2,
    const float* __restrict__ wT,
    float* __restrict__ out,
    float* __restrict__ aout) {
  __shared__ int sIdx[8 * 32];

  const int lane = threadIdx.x & 31;
  const int wv   = threadIdx.x >> 5;
  const int sel  = lane >> 4;       // half-wave select
  const int col  = lane & 15;
  const int g    = blockIdx.x * 8 + wv;   // wave-pair id, 0..4095
  const int t0   = g * 2;                 // first tile of the pair
  const int bimg = t0 >> 8;               // batch image (pairs never straddle)
  const int posb = (t0 & 255) << 4;       // base position of tile0

  // ---- A tiles: 2 x (16 positions x 128 channels), f32 -> f16, WMMA A layout
  const float* xp = x + (size_t)bimg * Dch * HW + (posb + col);
  v16h A0[4], A1[4];
#pragma unroll
  for (int t = 0; t < 4; ++t) {
    const int d0 = t * 32 + sel * 8;
    const int d1 = t * 32 + 16 + sel * 8;
#pragma unroll
    for (int h = 0; h < 8; ++h) {
      A0[t][h]     = (_Float16)xp[(size_t)(d0 + h) * HW];
      A0[t][8 + h] = (_Float16)xp[(size_t)(d1 + h) * HW];
      A1[t][h]     = (_Float16)xp[(size_t)(d0 + h) * HW + 16];
      A1[t][8 + h] = (_Float16)xp[(size_t)(d1 + h) * HW + 16];
    }
  }

  float bV0[8], bV1[8];
  int   bI0[8], bI1[8];
#pragma unroll
  for (int r = 0; r < 8; ++r) {
    bV0[r] = -3.4e38f; bI0[r] = 0;
    bV1[r] = -3.4e38f; bI1[r] = 0;
  }

  const v16h* Bp = (const v16h*)packed;
  v16h Ba[4], Bb[4];
  load_btile(Bp, 0, lane, Ba);
  float c0a = ne2[col];

  for (int c = 0; c < 64; c += 2) {
    // prefetch c+1 into the other buffer, compute c
    load_btile(Bp, c + 1, lane, Bb);
    const float c0b = ne2[(c + 1) * 16 + col];
    mma_step(A0, A1, Ba, c0a, c * 16 + col, bV0, bI0, bV1, bI1);

    // prefetch c+2 (wraps harmlessly on last iteration), compute c+1
    load_btile(Bp, (c + 2) & 63, lane, Ba);
    c0a = ne2[(((c + 2) & 63) * 16) + col];
    mma_step(A0, A1, Bb, c0b, (c + 1) * 16 + col, bV0, bI0, bV1, bI1);
  }

  // butterfly max-reduce over the 16 lanes (N dim), first-index tie-break
#pragma unroll
  for (int msk = 1; msk <= 8; msk <<= 1) {
#pragma unroll
    for (int r = 0; r < 8; ++r) {
      float ov; int oi;
      ov = __shfl_xor(bV0[r], msk, 32);
      oi = __shfl_xor(bI0[r], msk, 32);
      if (ov > bV0[r] || (ov == bV0[r] && oi < bI0[r])) { bV0[r] = ov; bI0[r] = oi; }
      ov = __shfl_xor(bV1[r], msk, 32);
      oi = __shfl_xor(bI1[r], msk, 32);
      if (ov > bV1[r] || (ov == bV1[r] && oi < bI1[r])) { bV1[r] = ov; bI1[r] = oi; }
    }
  }

  // exchange winners through per-wave LDS: row layout [wv][u*16 + m]
  if (col == 0) {
#pragma unroll
    for (int r = 0; r < 8; ++r) {
      sIdx[wv * 32 + sel * 8 + r]      = bI0[r];
      sIdx[wv * 32 + 16 + sel * 8 + r] = bI1[r];
    }
  }
  const int myIdx0 = sIdx[wv * 32 + col];
  const int myIdx1 = sIdx[wv * 32 + 16 + col];

  // argmin map: lanes 0-15 cover tile0 rows, lanes 16-31 cover tile1 rows
  aout[(size_t)bimg * HW + posb + lane] = (float)(sel ? myIdx1 : myIdx0);

  // gather winning f32 codebook rows (W^T, contiguous), scatter to NCHW out
#pragma unroll
  for (int u = 0; u < 2; ++u) {
    const int idx  = u ? myIdx1 : myIdx0;
    const int posl = posb + u * 16 + col;
    float* op = out + (size_t)bimg * Dch * HW + posl;
    const float4* wr = (const float4*)(wT + (size_t)idx * Dch + sel * 64);
#pragma unroll
    for (int j = 0; j < 16; ++j) {
      const float4 v = wr[j];
      const int d0 = sel * 64 + j * 4;
      op[(size_t)(d0 + 0) * HW] = v.x;
      op[(size_t)(d0 + 1) * HW] = v.y;
      op[(size_t)(d0 + 2) * HW] = v.z;
      op[(size_t)(d0 + 3) * HW] = v.w;
    }
  }
}

extern "C" void kernel_launch(void* const* d_in, const int* in_sizes, int n_in,
                              void* d_out, int out_size, void* d_ws, size_t ws_size,
                              hipStream_t stream) {
  (void)in_sizes; (void)n_in; (void)out_size; (void)ws_size;
  const float* x = (const float*)d_in[0];
  const float* w = (const float*)d_in[1];

  float* out  = (float*)d_out;
  float* aout = out + (size_t)NB * Dch * HW;   // argmin (as float) after result

  char* ws = (char*)d_ws;
  _Float16* packed = (_Float16*)(ws + WS_PACKED_OFF);
  float*    ne2    = (float*)(ws + WS_NE2_OFF);
  float*    wT     = (float*)(ws + WS_WT_OFF);

  vq_prep_kernel<<<131072 / 256, 256, 0, stream>>>(w, packed, ne2, wT);
  vq_main_kernel<<<NTILE / 16, 256, 0, stream>>>(x, packed, ne2, wT, out, aout);
}
